// SplineCouplingLayer_4758823764211
// MI455X (gfx1250) — matrix operand order repaired and checked by reading, not compiled, not run on hardware
//
#include <hip/hip_runtime.h>
#include <hip/hip_bf16.h>
#include <math.h>

// ---------------- problem constants ----------------
#define BOUND  5.0f
#define BM     64          // batch rows per block
#define NT     256         // threads per block (8 waves of 32)
#define NBLK   (32768/BM)  // 512 blocks

// ws layout (ushort elements): bf16 weights
#define W1ELEMS (512*64)        // W1 slice: only K=0..63 matter (mask)
#define W2ELEMS (512*512)
#define W3ROWS  1856            // W3 rows 1856..3711 (dims 64..127) = 116*16
#define W3ELEMS (W3ROWS*512)
#define W1OFF   0
#define W2OFF   (W1OFF+W1ELEMS)
#define W3OFF   (W2OFF+W2ELEMS)
#define WSTOT   (W3OFF+W3ELEMS)

// LDS layout (bytes)
#define ZF_OFF   0
#define ZF_BYTES (64*128*4)          // z tile f32
#define ZA_OFF   (ZF_OFF+ZF_BYTES)
#define ZA_PITCH 72
#define ZA_BYTES (64*ZA_PITCH*2)     // masked z tile bf16
#define H1_OFF   (ZA_OFF+ZA_BYTES)
#define H_PITCH  520
#define H_BYTES  (64*H_PITCH*2)      // hidden layer bf16
#define H2_OFF   (H1_OFF+H_BYTES)
#define PC_OFF   (H2_OFF+H_BYTES)
#define PC_PITCH 468
#define PC_BYTES (64*PC_PITCH*4)     // p chunk f32 (64 x 464 used)
#define LA_OFF   (PC_OFF+PC_BYTES)
#define SMEM_BYTES (LA_OFF + 64*4)   // + per-row logdet accum = 295168 B

typedef __attribute__((ext_vector_type(16))) __bf16 bf16x16;
typedef __attribute__((ext_vector_type(8)))  float  v8f;

struct B32B { uint4 lo, hi; };

__device__ __forceinline__ unsigned short f2bf(float f) {
    unsigned u = __float_as_uint(f);
    unsigned r = u + 0x7FFFu + ((u >> 16) & 1u);   // RNE
    return (unsigned short)(r >> 16);
}
__device__ __forceinline__ float fin0(float v) {
    return (isnan(v) || isinf(v)) ? 0.0f : v;
}
__device__ __forceinline__ v8f vzero() {
    v8f z;
    #pragma unroll
    for (int i = 0; i < 8; ++i) z[i] = 0.0f;
    return z;
}

// A fragment (16x32 bf16): lane<16 row M=lane, K={k0..k0+7, k0+16..k0+23};
// lane>=16 row M=lane-16, +8 on each K half.
__device__ __forceinline__ bf16x16 ldA(const unsigned short* base, int pitch,
                                       int m0, int k0, int lane) {
    int sel = (lane >> 4) << 3;                    // 0 or 8
    const unsigned short* p = base + (m0 + (lane & 15)) * pitch + k0;
    B32B r;
    r.lo = *(const uint4*)(p + sel);
    r.hi = *(const uint4*)(p + 16 + sel);
    return __builtin_bit_cast(bf16x16, r);
}
// B fragment (32x16 bf16): lane holds column N=lane%16, K={k0..k0+7, k0+16..k0+23}
// (lanes 0-15), +8 on each K half (lanes 16-31). Row-major [N][K] weights.
__device__ __forceinline__ bf16x16 ldB(const unsigned short* base, int pitch,
                                       int n0, int k0, int lane) {
    int col = k0 + ((lane >> 4) << 3);
    const unsigned short* p = base + (size_t)(n0 + (lane & 15)) * pitch + col;
    B32B r;
    r.lo = *(const uint4*)(p);
    r.hi = *(const uint4*)(p + 16);
    return __builtin_bit_cast(bf16x16, r);
}

__device__ __forceinline__ v8f wmma_bf16(bf16x16 a, bf16x16 b, v8f c) {
    return __builtin_amdgcn_wmma_f32_16x16x32_bf16(false, a, false, b,
                                                   (short)0, c, false, false);
}

// softmax -> min-width mix -> cumsum knots in [-B, B]
__device__ __forceinline__ void knots(const float* pp, const float* bb, float* cw) {
    float u[10]; float mx = -3.4e38f;
    #pragma unroll
    for (int j = 0; j < 10; ++j) { float t = pp[j] + bb[j]; u[j] = t; mx = fmaxf(mx, t); }
    float s = 0.f;
    #pragma unroll
    for (int j = 0; j < 10; ++j) { float e = __expf(u[j] - mx); u[j] = e; s += e; }
    float inv = 1.f / s, cs = 0.f;
    cw[0] = -BOUND;
    #pragma unroll
    for (int j = 0; j < 10; ++j) {
        float wj = 0.001f + 0.99f * u[j] * inv;
        cs += wj;
        cw[j + 1] = 2.f * BOUND * cs - BOUND;
    }
    cw[10] = BOUND;
}

// ---------------- weight conversion kernel ----------------
__global__ void prep_weights(const float* __restrict__ W1,
                             const float* __restrict__ W2,
                             const float* __restrict__ W3,
                             unsigned short* __restrict__ ws) {
    int i = blockIdx.x * blockDim.x + threadIdx.x;
    if (i < W1ELEMS) {
        int n = i >> 6, k = i & 63;                       // [512][64] <- W1[n][k], k<64
        ws[W1OFF + i] = f2bf(W1[n * 128 + k]);
    } else if (i < W1ELEMS + W2ELEMS) {
        int j = i - W1ELEMS;
        ws[W2OFF + j] = f2bf(W2[j]);
    } else if (i < WSTOT) {
        int j = i - (W1ELEMS + W2ELEMS);
        int r = j >> 9, k = j & 511;                      // W3 rows 1856..3711
        ws[W3OFF + j] = f2bf(W3[(size_t)(1856 + r) * 512 + k]);
    }
}

// ---------------- fused MLP + spline kernel ----------------
__global__ void __launch_bounds__(NT, 1)
spline_fused(const float* __restrict__ z,
             const unsigned short* __restrict__ wsbf,
             const float* __restrict__ b1,
             const float* __restrict__ b2,
             const float* __restrict__ b3,
             float* __restrict__ xout,
             float* __restrict__ ldout) {
    extern __shared__ unsigned char smem[];
    float*          zF  = (float*)(smem + ZF_OFF);
    unsigned short* zA  = (unsigned short*)(smem + ZA_OFF);
    unsigned short* h1  = (unsigned short*)(smem + H1_OFF);
    unsigned short* h2  = (unsigned short*)(smem + H2_OFF);
    float*          pch = (float*)(smem + PC_OFF);
    float*          lac = (float*)(smem + LA_OFF);

    const unsigned short* w1bf = wsbf + W1OFF;
    const unsigned short* w2bf = wsbf + W2OFF;
    const unsigned short* w3bf = wsbf + W3OFF;

    const int tid  = threadIdx.x;
    const int lane = tid & 31;
    const int wv   = tid >> 5;
    const int row0 = blockIdx.x * BM;

    // ---- stage z tile ----
    for (int i = tid; i < 64 * 32; i += NT) {
        int r = i >> 5, c = i & 31;
        ((float4*)(zF + r * 128))[c] = ((const float4*)(z + (size_t)(row0 + r) * 128))[c];
    }
    __syncthreads();
    if (tid < 64) lac[tid] = 0.0f;
    for (int i = tid; i < 64 * 64; i += NT) {   // masked bf16 A-tile (cols 0..63)
        int r = i >> 6, c = i & 63;
        zA[r * ZA_PITCH + c] = f2bf(zF[r * 128 + c]);
    }
    for (int i = tid; i < 64 * 64; i += NT) {   // passthrough half of x
        int r = i >> 6, c = i & 63;
        xout[(size_t)(row0 + r) * 128 + c] = fin0(zF[r * 128 + c]);
    }
    __syncthreads();

    const int N  = lane & 15;
    const int mh = (lane >> 4) << 3;

    // ---- GEMM1: h1 = relu(zA @ W1s^T + b1)  (64x512, K=64) ----
    // Each wave: 4 M-tiles x 4 N-tiles = 16 independent accumulators.
    {
        v8f acc[16];
        #pragma unroll
        for (int i = 0; i < 16; ++i) acc[i] = vzero();
        for (int k0 = 0; k0 < 64; k0 += 32) {
            bf16x16 a[4], b[4];
            #pragma unroll
            for (int mi = 0; mi < 4; ++mi) a[mi] = ldA(zA, ZA_PITCH, mi * 16, k0, lane);
            #pragma unroll
            for (int nj = 0; nj < 4; ++nj) b[nj] = ldB(w1bf, 64, (wv * 4 + nj) * 16, k0, lane);
            #pragma unroll
            for (int mi = 0; mi < 4; ++mi)
                #pragma unroll
                for (int nj = 0; nj < 4; ++nj)
                    acc[mi * 4 + nj] = wmma_bf16(a[mi], b[nj], acc[mi * 4 + nj]);
        }
        #pragma unroll
        for (int nj = 0; nj < 4; ++nj) {
            int n0 = (wv * 4 + nj) * 16;
            float bias = b1[n0 + N];
            #pragma unroll
            for (int mi = 0; mi < 4; ++mi) {
                v8f c = acc[mi * 4 + nj];
                #pragma unroll
                for (int j = 0; j < 8; ++j) {
                    float v = c[j] + bias; v = v > 0.f ? v : 0.f;
                    h1[(mi * 16 + mh + j) * H_PITCH + n0 + N] = f2bf(v);
                }
            }
        }
    }
    __syncthreads();

    // ---- GEMM2: h2 = relu(h1 @ W2^T + b2)  (64x512, K=512) ----
    {
        v8f acc[16];
        #pragma unroll
        for (int i = 0; i < 16; ++i) acc[i] = vzero();
        for (int k0 = 0; k0 < 512; k0 += 32) {
            bf16x16 a[4], b[4];
            #pragma unroll
            for (int mi = 0; mi < 4; ++mi) a[mi] = ldA(h1, H_PITCH, mi * 16, k0, lane);
            #pragma unroll
            for (int nj = 0; nj < 4; ++nj) b[nj] = ldB(w2bf, 512, (wv * 4 + nj) * 16, k0, lane);
            #pragma unroll
            for (int mi = 0; mi < 4; ++mi)
                #pragma unroll
                for (int nj = 0; nj < 4; ++nj)
                    acc[mi * 4 + nj] = wmma_bf16(a[mi], b[nj], acc[mi * 4 + nj]);
        }
        #pragma unroll
        for (int nj = 0; nj < 4; ++nj) {
            int n0 = (wv * 4 + nj) * 16;
            float bias = b2[n0 + N];
            #pragma unroll
            for (int mi = 0; mi < 4; ++mi) {
                v8f c = acc[mi * 4 + nj];
                #pragma unroll
                for (int j = 0; j < 8; ++j) {
                    float v = c[j] + bias; v = v > 0.f ? v : 0.f;
                    h2[(mi * 16 + mh + j) * H_PITCH + n0 + N] = f2bf(v);
                }
            }
        }
    }
    __syncthreads();

    // ---- GEMM3 (chunked over 4 x 464 cols) + fused spline ----
    // 29 N-tiles per chunk: wave w owns tiles {w, w+8, w+16, w+24}; out-of-range
    // tiles are clamped (redundant compute) and their stores skipped.
    for (int chk = 0; chk < 4; ++chk) {
        {
            v8f acc[16];
            #pragma unroll
            for (int i = 0; i < 16; ++i) acc[i] = vzero();
            int n0l[4]; bool val[4];
            #pragma unroll
            for (int g = 0; g < 4; ++g) {
                int nt = wv + 8 * g;
                val[g] = nt < 29;
                n0l[g] = (val[g] ? nt : 28) * 16;
            }
            for (int k0 = 0; k0 < 512; k0 += 32) {
                bf16x16 a[4], b[4];
                #pragma unroll
                for (int mi = 0; mi < 4; ++mi) a[mi] = ldA(h2, H_PITCH, mi * 16, k0, lane);
                #pragma unroll
                for (int g = 0; g < 4; ++g) b[g] = ldB(w3bf, 512, chk * 464 + n0l[g], k0, lane);
                #pragma unroll
                for (int mi = 0; mi < 4; ++mi)
                    #pragma unroll
                    for (int g = 0; g < 4; ++g)
                        acc[mi * 4 + g] = wmma_bf16(a[mi], b[g], acc[mi * 4 + g]);
            }
            #pragma unroll
            for (int g = 0; g < 4; ++g) {
                if (!val[g]) continue;
                #pragma unroll
                for (int mi = 0; mi < 4; ++mi) {
                    v8f c = acc[mi * 4 + g];
                    #pragma unroll
                    for (int j = 0; j < 8; ++j)
                        pch[(mi * 16 + mh + j) * PC_PITCH + n0l[g] + N] = c[j];
                }
            }
        }
        __syncthreads();

        // spline for 16 transformed dims of this chunk, 64 rows
        for (int task = tid; task < 64 * 16; task += NT) {
            int r  = task & 63;
            int dl = task >> 6;
            int tg = chk * 16 + dl;                   // dim index within TIDX half
            const float* pp = pch + r * PC_PITCH + dl * 29;
            const float* bb = b3 + (size_t)(64 + tg) * 29;

            float x = zF[r * 128 + 64 + tg];

            float cw[11], chh[11], dk[11];
            knots(pp, bb, cw);
            knots(pp + 10, bb + 10, chh);
            dk[0] = 1.0f; dk[10] = 1.0f;
            #pragma unroll
            for (int j = 0; j < 9; ++j) {
                float u = pp[20 + j] + bb[20 + j];
                float sp = (u > 20.f) ? u : logf(1.0f + __expf(u));   // softplus
                dk[j + 1] = 0.001f + sp;
            }

            float xc = fminf(fmaxf(x, -BOUND), BOUND);
            int b = 0;
            #pragma unroll
            for (int i = 1; i <= 10; ++i) b += (cw[i] <= xc) ? 1 : 0;
            if (b > 9) b = 9;

            float cwk = cw[b],  wk  = cw[b + 1] - cw[b];
            float chk2 = chh[b], hk = chh[b + 1] - chh[b];
            float dkb = dk[b], dk1 = dk[b + 1];
            float theta = (xc - cwk) / wk;
            float s  = hk / wk;
            float tm = theta * (1.0f - theta);
            float den = s + (dkb + dk1 - 2.0f * s) * tm;
            float num = hk * (s * theta * theta + dkb * tm);
            float y  = chk2 + num / den;
            float omt = 1.0f - theta;
            float ld = logf(s * s * (dk1 * theta * theta + 2.0f * s * tm + dkb * omt * omt))
                       - 2.0f * logf(den);
            bool inside = (x >= -BOUND) && (x <= BOUND);
            y  = inside ? y  : x;
            ld = inside ? ld : 0.0f;

            xout[(size_t)(row0 + r) * 128 + 64 + tg] = fin0(y);
            atomicAdd(&lac[r], ld);
        }
        __syncthreads();
    }

    if (tid < 64) ldout[row0 + tid] = fin0(lac[tid]);
}

// ---------------- launch ----------------
extern "C" void kernel_launch(void* const* d_in, const int* in_sizes, int n_in,
                              void* d_out, int out_size, void* d_ws, size_t ws_size,
                              hipStream_t stream) {
    const float* z  = (const float*)d_in[0];
    const float* W1 = (const float*)d_in[1];
    const float* b1 = (const float*)d_in[2];
    const float* W2 = (const float*)d_in[3];
    const float* b2 = (const float*)d_in[4];
    const float* W3 = (const float*)d_in[5];
    const float* b3 = (const float*)d_in[6];

    unsigned short* wsbf = (unsigned short*)d_ws;
    float* xout  = (float*)d_out;                 // 32768*128 floats
    float* ldout = xout + (size_t)32768 * 128;    // then 32768 logdets

    prep_weights<<<(WSTOT + 255) / 256, 256, 0, stream>>>(W1, W2, W3, wsbf);

    hipFuncSetAttribute((const void*)spline_fused,
                        hipFuncAttributeMaxDynamicSharedMemorySize, SMEM_BYTES);
    spline_fused<<<NBLK, NT, SMEM_BYTES, stream>>>(z, wsbf, b1, b2, b3, xout, ldout);
}
